// BottleneckBlock_28870770164307
// MI455X (gfx1250) — matrix operand order repaired and checked by reading, not compiled, not run on hardware
//
#include <hip/hip_runtime.h>
#include <hip/hip_bf16.h>

typedef __attribute__((ext_vector_type(16))) _Float16 v16h;
typedef __attribute__((ext_vector_type(8)))  float    v8f;

// Problem constants
#define BATCH 2
#define TT 64
#define CH 512
#define HW 576            // 24*24
#define SEQS (BATCH*HW)   // 1152
#define DIM 256
#define NQKV 768
#define HEADS 8
#define HD 32
#define MROWS (SEQS*TT)   // 73728
#define QSCALE_F 0.17677669529663687f  // 32^-0.5

#define APAD 264          // K-panel row stride in halfs (bank-conflict-free, 16B aligned)

union Frag { unsigned int u[8]; _Float16 e[16]; v16h v; };

// ---------------- weight f32 -> f16 ----------------
__global__ __launch_bounds__(256) void prep_weights(const float* __restrict__ qkv_w,
                                                    const float* __restrict__ proj_w,
                                                    _Float16* __restrict__ Wqkv,
                                                    _Float16* __restrict__ Wp) {
  int idx = blockIdx.x * 256 + threadIdx.x;
  if (idx < NQKV * DIM) Wqkv[idx] = (_Float16)qkv_w[idx];
  int i2 = idx - NQKV * DIM;
  if (i2 >= 0 && i2 < DIM * DIM) Wp[i2] = (_Float16)proj_w[i2];
}

// ---------------- passthrough half: out[:, :, :256] = x[:, :, :256] ----------------
__global__ __launch_bounds__(256) void copy_a(const float* __restrict__ x,
                                              float* __restrict__ out) {
  size_t idx = (size_t)blockIdx.x * 256 + threadIdx.x; // over B*T*256*HW
  int hw = (int)(idx % HW); size_t r = idx / HW;
  int c  = (int)(r % DIM);  size_t bt = r / DIM;
  size_t o = (bt * CH + c) * HW + hw;
  out[o] = x[o];
}

// ---------------- pack: x[b,t,256+c,hw] -> Xp[(b*HW+hw)*T + t][c] (f16) ----------------
__global__ __launch_bounds__(256) void pack_b(const float* __restrict__ x,
                                              _Float16* __restrict__ Xp) {
  __shared__ __align__(16) float tile[32][65];
  int bid = blockIdx.x;
  int hwT = bid % 9; int rem = bid / 9;
  int cT  = rem % 8; int bt  = rem / 8;     // bt = b*T + t
  int b = bt >> 6, t = bt & 63;
  int c0 = cT * 32, hw0 = hwT * 64;
  int tid = threadIdx.x;
  for (int i = tid; i < 2048; i += 256) {   // coalesced over hw
    int cl = i >> 6, hwl = i & 63;
    tile[cl][hwl] = x[((size_t)bt * CH + DIM + c0 + cl) * HW + hw0 + hwl];
  }
  __syncthreads();
  for (int i = tid; i < 2048; i += 256) {   // contiguous over c
    int hwl = i >> 5, cl = i & 31;
    int s = b * HW + hw0 + hwl;
    Xp[((size_t)s * TT + t) * DIM + c0 + cl] = (_Float16)tile[cl][hwl];
  }
}

// ---------------- WMMA GEMM: out[M x N] = A[M x 256] @ Bw[N x 256]^T + bias ----------
// Full-K panels staged once into 99 KB of dynamic LDS (320 KB/WGP on CDNA5 ->
// 3 workgroups resident), then a sync-free k-loop of pure ds_load_b128 + v_wmma.
template <bool OUTF16, bool QSCALE>
__global__ __launch_bounds__(256) void gemm_wmma(const _Float16* __restrict__ A,
                                                 const _Float16* __restrict__ Bw,
                                                 const float* __restrict__ bias,
                                                 _Float16* __restrict__ outh,
                                                 float* __restrict__ outf,
                                                 int N) {
  extern __shared__ __align__(16) _Float16 smem[];
  _Float16* Asm = smem;                       // 128 x APAD halfs (A panel, K=256)
  _Float16* Bsm = smem + 128 * APAD;          // 64  x APAD halfs (B panel, [n][k])
  const int tid  = threadIdx.x;
  const int nblk = N >> 6;
  const int bn = blockIdx.x % nblk, bm = blockIdx.x / nblk;
  const int m0 = bm * 128, n0 = bn * 64;
  const int wave = tid >> 5, lane = tid & 31;
  const int wm = wave >> 1, wn = wave & 1;
  const int half = lane >> 4, l15 = lane & 15;

  // stage A panel: 128 rows x 256 halfs, 16B chunks, coalesced
  #pragma unroll
  for (int it = 0; it < 16; ++it) {
    int i = tid + it * 256;
    int row = i >> 5, c8 = (i & 31) << 3;
    *(uint4*)(Asm + row * APAD + c8) = *(const uint4*)(A + (size_t)(m0 + row) * 256 + c8);
  }
  // stage B panel: 64 rows x 256 halfs
  #pragma unroll
  for (int it = 0; it < 8; ++it) {
    int i = tid + it * 256;
    int row = i >> 5, c8 = (i & 31) << 3;
    *(uint4*)(Bsm + row * APAD + c8) = *(const uint4*)(Bw + (size_t)(n0 + row) * 256 + c8);
  }
  __syncthreads();

  v8f acc[2][2] = {};
  for (int kk = 0; kk < 256; kk += 32) {
    const int kb = kk >> 1;   // uint offset into a row
    Frag af[2], bf[2];
    #pragma unroll
    for (int mt = 0; mt < 2; ++mt) {
      const unsigned int* arow =
          (const unsigned int*)(Asm + (wm * 32 + mt * 16 + l15) * APAD) + kb;
      #pragma unroll
      for (int v = 0; v < 8; ++v)
        af[mt].u[v] = arow[half * 8 + v];
    }
    #pragma unroll
    for (int nt = 0; nt < 2; ++nt) {
      const unsigned int* brow =
          (const unsigned int*)(Bsm + (wn * 32 + nt * 16 + l15) * APAD) + kb;
      #pragma unroll
      for (int v = 0; v < 8; ++v)
        bf[nt].u[v] = brow[half * 8 + v];
    }
    #pragma unroll
    for (int mt = 0; mt < 2; ++mt)
      #pragma unroll
      for (int nt = 0; nt < 2; ++nt)
        acc[mt][nt] = __builtin_amdgcn_wmma_f32_16x16x32_f16(
            false, af[mt].v, false, bf[nt].v, (short)0, acc[mt][nt], false, false);
  }

  // epilogue: bias (+ q-scale on first 256 cols for QKV)
  #pragma unroll
  for (int mt = 0; mt < 2; ++mt)
    #pragma unroll
    for (int nt = 0; nt < 2; ++nt) {
      int gn = n0 + wn * 32 + nt * 16 + l15;
      float bv = bias[gn];
      float sc = (QSCALE && gn < DIM) ? QSCALE_F : 1.0f;
      #pragma unroll
      for (int r = 0; r < 8; ++r) {
        int gm = m0 + wm * 32 + mt * 16 + r + half * 8;
        float val = (acc[mt][nt][r] + bv) * sc;
        size_t o = (size_t)gm * N + gn;
        if (OUTF16) outh[o] = (_Float16)val; else outf[o] = val;
      }
    }
}

// ---------------- attention per (sequence, head) ----------------
__global__ __launch_bounds__(128) void attn_kernel(const _Float16* __restrict__ QKVp,
                                                   const float* __restrict__ rpb,
                                                   _Float16* __restrict__ Op) {
  __shared__ __align__(16) _Float16 Qs[64][40];
  __shared__ __align__(16) _Float16 Ks[64][40];
  __shared__ __align__(16) _Float16 Vs[64][40];
  __shared__ float Sc[64][65];
  __shared__ _Float16 Ps[64][72];
  __shared__ float rl[128];
  const int tid = threadIdx.x;
  const int h = blockIdx.x & 7;
  const int s = blockIdx.x >> 3;
  const size_t base = (size_t)s * TT * NQKV + h * HD;
  for (int i = tid; i < 256; i += 128) {
    int t = i >> 2, d8 = (i & 3) << 3;
    const _Float16* src = QKVp + base + (size_t)t * NQKV + d8;
    *(uint4*)&Qs[t][d8] = *(const uint4*)(src);
    *(uint4*)&Ks[t][d8] = *(const uint4*)(src + DIM);
    *(uint4*)&Vs[t][d8] = *(const uint4*)(src + 2 * DIM);
  }
  if (tid < 127) rl[tid] = rpb[tid * HEADS + h];
  __syncthreads();
  const int wave = tid >> 5, lane = tid & 31, half = lane >> 4, l15 = lane & 15;
  const int m0 = wave * 16;
  // scores: S = Q K^T  (wave handles 16-row strip, 4 WMMAs across j).
  // Raw scores only -- bias + causal mask are folded into the softmax pass,
  // keeping the WMMA region free of scalar bias traffic.
  Frag aq;
  {
    const unsigned int* qrow = (const unsigned int*)&Qs[m0 + l15][0];
    #pragma unroll
    for (int v = 0; v < 8; ++v) aq.u[v] = qrow[half * 8 + v];
  }
  #pragma unroll
  for (int jt = 0; jt < 4; ++jt) {
    Frag bk;
    const unsigned int* krow = (const unsigned int*)&Ks[jt * 16 + l15][0];
    #pragma unroll
    for (int v = 0; v < 8; ++v) bk.u[v] = krow[half * 8 + v];
    v8f accs = {};
    accs = __builtin_amdgcn_wmma_f32_16x16x32_f16(false, aq.v, false, bk.v,
                                                  (short)0, accs, false, false);
    #pragma unroll
    for (int r = 0; r < 8; ++r)
      Sc[m0 + r + half * 8][jt * 16 + l15] = accs[r];
  }
  __syncthreads();
  // softmax with fused rpb bias + causal mask:
  // 2 lanes per row (64 rows, all 128 threads active), pair-combine via shfl.
  // rl index = j - row + 63 increments linearly -> base pointer + imm offsets.
  {
    int row = tid >> 1, seg = tid & 1;
    int jbase = seg * 32;
    float* sr = &Sc[row][0] + jbase;
    const float* bp = &rl[jbase - row + 63];   // in [0,126] for all accessed jj
    float m = -1.0e30f;
    #pragma unroll 8
    for (int jj = 0; jj < 32; ++jj) {
      int j = jbase + jj;
      float v = (j > row) ? -1.0e30f : (sr[jj] + bp[jj]);
      sr[jj] = v;
      m = fmaxf(m, v);
    }
    m = fmaxf(m, __shfl_xor(m, 1));
    float sum = 0.f;
    #pragma unroll 8
    for (int jj = 0; jj < 32; ++jj) { float e = __expf(sr[jj] - m); sr[jj] = e; sum += e; }
    sum += __shfl_xor(sum, 1);
    float inv = 1.0f / sum;
    _Float16* pr = &Ps[row][0] + jbase;
    #pragma unroll 8
    for (int jj = 0; jj < 32; ++jj) pr[jj] = (_Float16)(sr[jj] * inv);
  }
  __syncthreads();
  // O = P @ V : two k-steps of 32, two 16-wide d tiles
  v8f acco[2] = {};
  #pragma unroll
  for (int ks = 0; ks < 2; ++ks) {
    Frag ap;
    const unsigned int* prow = (const unsigned int*)&Ps[m0 + l15][0] + ks * 16;
    #pragma unroll
    for (int v = 0; v < 8; ++v) ap.u[v] = prow[half * 8 + v];
    #pragma unroll
    for (int nt = 0; nt < 2; ++nt) {
      Frag bv;
      #pragma unroll
      for (int v = 0; v < 8; ++v) {
        int k = ks * 32 + half * 16 + 2 * v;
        int n = nt * 16 + l15;
        bv.e[2 * v]     = Vs[k][n];
        bv.e[2 * v + 1] = Vs[k + 1][n];
      }
      acco[nt] = __builtin_amdgcn_wmma_f32_16x16x32_f16(false, ap.v, false, bv.v,
                                                        (short)0, acco[nt], false, false);
    }
  }
  #pragma unroll
  for (int nt = 0; nt < 2; ++nt)
    #pragma unroll
    for (int r = 0; r < 8; ++r) {
      int t = m0 + r + half * 8;
      int c = h * HD + nt * 16 + l15;
      Op[((size_t)s * TT + t) * DIM + c] = (_Float16)acco[nt][r];
    }
}

// ---------------- unpack: Yp[s][t][c] -> out[b,t,256+c,hw] ----------------
__global__ __launch_bounds__(256) void unpack_b(const float* __restrict__ Yp,
                                                float* __restrict__ out) {
  __shared__ __align__(16) float tile[64][33];
  int bid = blockIdx.x;
  int hwT = bid % 9; int rem = bid / 9;
  int cT  = rem % 8; int bt  = rem / 8;
  int b = bt >> 6, t = bt & 63;
  int c0 = cT * 32, hw0 = hwT * 64;
  int tid = threadIdx.x;
  for (int i = tid; i < 2048; i += 256) {   // contiguous over c
    int hwl = i >> 5, cl = i & 31;
    int s = b * HW + hw0 + hwl;
    tile[hwl][cl] = Yp[((size_t)s * TT + t) * DIM + c0 + cl];
  }
  __syncthreads();
  for (int i = tid; i < 2048; i += 256) {   // coalesced over hw
    int cl = i >> 6, hwl = i & 63;
    out[((size_t)bt * CH + DIM + c0 + cl) * HW + hw0 + hwl] = tile[hwl][cl];
  }
}

extern "C" void kernel_launch(void* const* d_in, const int* in_sizes, int n_in,
                              void* d_out, int out_size, void* d_ws, size_t ws_size,
                              hipStream_t stream) {
  const float* x      = (const float*)d_in[0];
  const float* rpb    = (const float*)d_in[1];
  const float* qkv_w  = (const float*)d_in[2];
  const float* qkv_b  = (const float*)d_in[3];
  const float* proj_w = (const float*)d_in[4];
  const float* proj_b = (const float*)d_in[5];
  float* out = (float*)d_out;

  // workspace layout (aliased across pipeline phases; every region is fully
  // written before it is read, so no state survives between calls)
  char* ws = (char*)d_ws;
  size_t off = 0;
  _Float16* Wqkv = (_Float16*)(ws + off); off += (size_t)NQKV * DIM * 2;   // 384 KB
  _Float16* Wp   = (_Float16*)(ws + off); off += (size_t)DIM * DIM * 2;    // 128 KB
  off = (off + 255) & ~(size_t)255;
  _Float16* Xp   = (_Float16*)(ws + off);                                  // 37.75 MB
  _Float16* Op   = Xp;                       // reuse: Xp dead after QKV GEMM
  off += (size_t)MROWS * DIM * 2;
  off = (off + 255) & ~(size_t)255;
  _Float16* QKVp = (_Float16*)(ws + off);                                  // 113 MB
  float*    Yp   = (float*)QKVp;             // reuse: QKVp dead after attention
  (void)ws_size; (void)in_sizes; (void)n_in; (void)out_size;

  const size_t gemm_smem = (size_t)(128 + 64) * APAD * sizeof(_Float16);   // ~99 KB

  prep_weights<<<(NQKV * DIM + DIM * DIM) / 256, 256, 0, stream>>>(qkv_w, proj_w, Wqkv, Wp);
  copy_a<<<(BATCH * TT * DIM * HW) / 256, 256, 0, stream>>>(x, out);
  pack_b<<<BATCH * TT * 8 * 9, 256, 0, stream>>>(x, Xp);
  gemm_wmma<true, true><<<(MROWS / 128) * (NQKV / 64), 256, gemm_smem, stream>>>(
      Xp, Wqkv, qkv_b, QKVp, nullptr, NQKV);
  attn_kernel<<<SEQS * HEADS, 128, 0, stream>>>(QKVp, rpb, Op);
  gemm_wmma<false, false><<<(MROWS / 128) * (DIM / 64), 256, gemm_smem, stream>>>(
      Op, Wp, proj_b, nullptr, Yp, DIM);
  unpack_b<<<BATCH * TT * 8 * 9, 256, 0, stream>>>(Yp, out);
}
